// UnquantizedMoELayer_31610959299085
// MI455X (gfx1250) — matrix operand order repaired and checked by reading, not compiled, not run on hardware
//
#include <hip/hip_runtime.h>
#include <hip/hip_bf16.h>

// ---------------------------------------------------------------------------
// MoE (top-2 of 8 experts, SwiGLU) for MI455X / gfx1250, wave32 + WMMA f16.
//
// T=2048, H=2048, I=5632, E=8, TOPK=2, renormalized routing.
//
// Roofline: routed FLOPs ~283 GFLOP; mandatory HBM traffic = fp32 weights
// ~1.11 GB -> ~48us @ 23.3 TB/s, so compute (f16 WMMA) and memory are nearly
// balanced. Strategy:
//   * compute only routed tokens (4x less math than the dense reference)
//   * read fp32 weights once, f32->f16 (v_cvt_pk_f16_f32) in regs into LDS
//   * SwiGLU intermediate kept f16: 46 MB -> resident in the 192 MB L2
//   * BM=64 token tiles halve per-expert weight re-streaming; 32x32 per-wave
//     tiles give 0.75 LDS fragment loads per v_wmma_f32_16x16x32_f16.
//
// Workspace (~46.2 MB of d_ws):
//   int   cnt[16] | int tmap[E*T] | float wslot[2*T] | f16 h_ws[2*T][I]
// ---------------------------------------------------------------------------

typedef __attribute__((ext_vector_type(16))) _Float16 v16h;
typedef __attribute__((ext_vector_type(8)))  _Float16 v8h;
typedef __attribute__((ext_vector_type(8)))  float    v8f;

constexpr int T_  = 2048;
constexpr int H_  = 2048;
constexpr int I_  = 5632;
constexpr int E_  = 8;
constexpr int BM  = 64;       // token rows per block
constexpr int BN  = 128;      // output cols per block
constexpr int BK  = 64;       // K step
constexpr int LDST = BK + 8;  // f16 LDS row stride (144B: skewed, 16B aligned)

// ---- f32 global -> f16 LDS, 8 elements (two b128 loads, one ds_store_b128)
__device__ __forceinline__ void cvt_store8(_Float16* __restrict__ dst,
                                           const float* __restrict__ src) {
  const float4 a = ((const float4*)src)[0];
  const float4 b = ((const float4*)src)[1];
  v8h o;
  o[0] = (_Float16)a.x; o[1] = (_Float16)a.y; o[2] = (_Float16)a.z; o[3] = (_Float16)a.w;
  o[4] = (_Float16)b.x; o[5] = (_Float16)b.y; o[6] = (_Float16)b.z; o[7] = (_Float16)b.w;
  *(v8h*)dst = o;
}

// ---- A fragment (16x32 f16, M x K). ISA 7.12.2: lane L -> row L%16,
// K in [h*8, h*8+8) U [16+h*8, 16+h*8+8), h = L/16.
__device__ __forceinline__ v16h frag_a(const _Float16* __restrict__ lds,
                                       int r0, int ko, int lane) {
  const int r = r0 + (lane & 15), h = lane >> 4;
  const _Float16* p = lds + r * LDST + ko + h * 8;
  v8h lo = *(const v8h*)p;
  v8h hi = *(const v8h*)(p + 16);
  return __builtin_shufflevector(lo, hi, 0,1,2,3,4,5,6,7,8,9,10,11,12,13,14,15);
}

// ---- B fragment (32x16 f16, K x N) with B supplied row-major [N][K]
// (weight rows; WMMA computes A * W^T = x @ w.T).
// lane L -> col L%16, K in [h*8, h*8+8) U [16+h*8, 16+h*8+8), h = L/16.
__device__ __forceinline__ v16h frag_b(const _Float16* __restrict__ lds,
                                       int r0, int ko, int lane) {
  const int r = r0 + (lane & 15), h = lane >> 4;
  const _Float16* p = lds + r * LDST + ko + h * 8;
  v8h lo = *(const v8h*)p;
  v8h hi = *(const v8h*)(p + 16);
  return __builtin_shufflevector(lo, hi, 0,1,2,3,4,5,6,7,8,9,10,11,12,13,14,15);
}

__device__ __forceinline__ v8f wmma(v16h a, v16h b, v8f c) {
  return __builtin_amdgcn_wmma_f32_16x16x32_f16(false, a, false, b,
                                                (short)0, c, false, false);
}

// ===========================================================================
// Kernel 0: zero the output accumulator and the expert counters.
// ===========================================================================
__global__ __launch_bounds__(256) void moe_zero(float* __restrict__ out,
                                                int* __restrict__ cnt) {
  const size_t i = (size_t)blockIdx.x * 256 + threadIdx.x;
  if (i < (size_t)T_ * H_) out[i] = 0.0f;
  if (i < E_) cnt[i] = 0;
}

// ===========================================================================
// Kernel 1: routing. softmax->top2->renorm collapses to sigmoid(l0-l1).
// Atomic-append (t*2+slot) to per-expert lists; final output is invariant to
// list order (rows independent; the two out-adds commute bitwise).
// ===========================================================================
__global__ __launch_bounds__(256) void moe_route(const float* __restrict__ gate,
                                                 int* __restrict__ cnt,
                                                 int* __restrict__ tmap,
                                                 float* __restrict__ wslot) {
  const int t = blockIdx.x * 256 + threadIdx.x;
  if (t >= T_) return;
  float l[E_];
#pragma unroll
  for (int i = 0; i < E_; ++i) l[i] = gate[t * E_ + i];
  int i0 = 0; float b0 = l[0];
#pragma unroll
  for (int i = 1; i < E_; ++i) if (l[i] > b0) { b0 = l[i]; i0 = i; }  // first max
  int i1 = (i0 == 0) ? 1 : 0; float b1 = l[i1];
#pragma unroll
  for (int i = 0; i < E_; ++i)
    if (i != i0 && l[i] > b1) { b1 = l[i]; i1 = i; }
  const float w0 = 1.0f / (1.0f + __expf(b1 - b0));  // = p0/(p0+p1)
  const float w1 = 1.0f - w0;
  const int p0 = atomicAdd(&cnt[i0], 1);
  tmap[i0 * T_ + p0] = t * 2 + 0;  wslot[t * 2 + 0] = w0;
  const int p1 = atomicAdd(&cnt[i1], 1);
  tmap[i1 * T_ + p1] = t * 2 + 1;  wslot[t * 2 + 1] = w1;
}

// ===========================================================================
// Kernel 2: GEMM1 + SwiGLU.  h = silu(x@Wg^T) * (x@Wu^T) in f16.
// Block: BM=64 x BN=128 (gate & up), 256 thr = 8 waves in 2(M)x4(N) grid,
// each wave owns a 32x32 gate tile + 32x32 up tile (8 f32 accumulators).
// 16 v_wmma per K-step.
// ===========================================================================
__global__ __launch_bounds__(256) void moe_gemm1(const float* __restrict__ x,
                                                 const float* __restrict__ w1,
                                                 const int* __restrict__ cnt,
                                                 const int* __restrict__ tmap,
                                                 _Float16* __restrict__ h_ws) {
  __shared__ __align__(16) _Float16 sX[BM * LDST];
  __shared__ __align__(16) _Float16 sG[BN * LDST];
  __shared__ __align__(16) _Float16 sU[BN * LDST];
  __shared__ int sEnt[BM];

  const int e = blockIdx.z;
  const int nTok = cnt[e];
  const int m0 = blockIdx.y * BM;
  if (m0 >= nTok) return;
  const int i0 = blockIdx.x * BN;

  const int tid = threadIdx.x, lane = tid & 31, wid = tid >> 5;
  const int m0w = (wid >> 2) * 32, n0w = (wid & 3) * 32;

  if (tid < BM) sEnt[tid] = (m0 + tid < nTok) ? tmap[e * T_ + m0 + tid] : 0;
  __syncthreads();

  const float* gB = w1 + ((size_t)e * 2 * I_ + i0) * H_;        // gate rows
  const float* uB = w1 + ((size_t)e * 2 * I_ + I_ + i0) * H_;   // up rows

  v8f accg[2][2] = {}, accu[2][2] = {};
  for (int kk = 0; kk < H_; kk += BK) {
#pragma unroll
    for (int it = 0; it < 2; ++it) {  // x tile: 64 rows x 64 f32, gathered
      const int idx = tid + it * 256, row = idx >> 3, c = (idx & 7) * 8;
      const float* s = x + (size_t)(sEnt[row] >> 1) * H_ + kk + c;
      cvt_store8(&sX[row * LDST + c], s);
    }
#pragma unroll
    for (int it = 0; it < 4; ++it) {  // gate+up tiles: 128 rows x 64 f32 each
      const int idx = tid + it * 256, row = idx >> 3, c = (idx & 7) * 8;
      const size_t off = (size_t)row * H_ + kk + c;
      cvt_store8(&sG[row * LDST + c], gB + off);
      cvt_store8(&sU[row * LDST + c], uB + off);
      if (kk + BK < H_) {  // global_prefetch_b8 hints for the next K-step
        __builtin_prefetch(gB + off + BK, 0, 3);
        __builtin_prefetch(uB + off + BK, 0, 3);
      }
    }
    __syncthreads();
#pragma unroll
    for (int s = 0; s < 2; ++s) {
      const int ko = s * 32;
      const v16h a0  = frag_a(sX, m0w,      ko, lane);
      const v16h a1  = frag_a(sX, m0w + 16, ko, lane);
      const v16h bg0 = frag_b(sG, n0w,      ko, lane);
      const v16h bg1 = frag_b(sG, n0w + 16, ko, lane);
      const v16h bu0 = frag_b(sU, n0w,      ko, lane);
      const v16h bu1 = frag_b(sU, n0w + 16, ko, lane);
      accg[0][0] = wmma(a0, bg0, accg[0][0]);
      accg[0][1] = wmma(a0, bg1, accg[0][1]);
      accg[1][0] = wmma(a1, bg0, accg[1][0]);
      accg[1][1] = wmma(a1, bg1, accg[1][1]);
      accu[0][0] = wmma(a0, bu0, accu[0][0]);
      accu[0][1] = wmma(a0, bu1, accu[0][1]);
      accu[1][0] = wmma(a1, bu0, accu[1][0]);
      accu[1][1] = wmma(a1, bu1, accu[1][1]);
    }
    __syncthreads();
  }

  // Epilogue: SwiGLU + f16 store. C/D: m = v + 8*(lane/16), n = lane%16.
  const int hh = lane >> 4, nn = lane & 15;
#pragma unroll
  for (int mi = 0; mi < 2; ++mi)
#pragma unroll
    for (int ni = 0; ni < 2; ++ni)
#pragma unroll
      for (int v = 0; v < 8; ++v) {
        const int ml = m0w + mi * 16 + v + 8 * hh;
        if (m0 + ml < nTok) {
          const float g = accg[mi][ni][v], u = accu[mi][ni][v];
          const float sig = 1.0f / (1.0f + __expf(-g));
          h_ws[(size_t)sEnt[ml] * I_ + i0 + n0w + ni * 16 + nn] =
              (_Float16)(g * sig * u);
        }
      }
}

// ===========================================================================
// Kernel 3: GEMM2.  out[t,:] += route_w * (h @ w2[e]^T).
// Same tiling; A = f16 h rows (pure b128 copy into LDS), 8 v_wmma per K-step.
// ===========================================================================
__global__ __launch_bounds__(256) void moe_gemm2(const _Float16* __restrict__ h_ws,
                                                 const float* __restrict__ w2,
                                                 const int* __restrict__ cnt,
                                                 const int* __restrict__ tmap,
                                                 const float* __restrict__ wslot,
                                                 float* __restrict__ out) {
  __shared__ __align__(16) _Float16 sH[BM * LDST];
  __shared__ __align__(16) _Float16 sW[BN * LDST];
  __shared__ int   sEnt[BM];
  __shared__ float sWt[BM];

  const int e = blockIdx.z;
  const int nTok = cnt[e];
  const int m0 = blockIdx.y * BM;
  if (m0 >= nTok) return;
  const int h0 = blockIdx.x * BN;

  const int tid = threadIdx.x, lane = tid & 31, wid = tid >> 5;
  const int m0w = (wid >> 2) * 32, n0w = (wid & 3) * 32;

  if (tid < BM) {
    const int en = (m0 + tid < nTok) ? tmap[e * T_ + m0 + tid] : 0;
    sEnt[tid] = en;
    sWt[tid]  = wslot[en];
  }
  __syncthreads();

  const float* wB = w2 + ((size_t)e * H_ + h0) * I_;

  v8f acc[2][2] = {};
  for (int kk = 0; kk < I_; kk += BK) {
#pragma unroll
    for (int it = 0; it < 2; ++it) {  // h tile: 64 rows x 64 f16, gathered
      const int idx = tid + it * 256, row = idx >> 3, c = (idx & 7) * 8;
      const _Float16* s = h_ws + (size_t)sEnt[row] * I_ + kk + c;
      *(v8h*)&sH[row * LDST + c] = *(const v8h*)s;
    }
#pragma unroll
    for (int it = 0; it < 4; ++it) {  // w2 tile: 128 rows x 64 f32
      const int idx = tid + it * 256, row = idx >> 3, c = (idx & 7) * 8;
      const size_t off = (size_t)row * I_ + kk + c;
      cvt_store8(&sW[row * LDST + c], wB + off);
      if (kk + BK < I_) __builtin_prefetch(wB + off + BK, 0, 3);
    }
    __syncthreads();
#pragma unroll
    for (int s = 0; s < 2; ++s) {
      const int ko = s * 32;
      const v16h a0 = frag_a(sH, m0w,      ko, lane);
      const v16h a1 = frag_a(sH, m0w + 16, ko, lane);
      const v16h b0 = frag_b(sW, n0w,      ko, lane);
      const v16h b1 = frag_b(sW, n0w + 16, ko, lane);
      acc[0][0] = wmma(a0, b0, acc[0][0]);
      acc[0][1] = wmma(a0, b1, acc[0][1]);
      acc[1][0] = wmma(a1, b0, acc[1][0]);
      acc[1][1] = wmma(a1, b1, acc[1][1]);
    }
    __syncthreads();
  }

  const int hh = lane >> 4, nn = lane & 15;
#pragma unroll
  for (int mi = 0; mi < 2; ++mi)
#pragma unroll
    for (int ni = 0; ni < 2; ++ni)
#pragma unroll
      for (int v = 0; v < 8; ++v) {
        const int ml = m0w + mi * 16 + v + 8 * hh;
        if (m0 + ml < nTok) {
          const int   t = sEnt[ml] >> 1;
          const float val = acc[mi][ni][v] * sWt[ml];
          atomicAdd(&out[(size_t)t * H_ + h0 + n0w + ni * 16 + nn], val);
        }
      }
}

// ===========================================================================
extern "C" void kernel_launch(void* const* d_in, const int* in_sizes, int n_in,
                              void* d_out, int out_size, void* d_ws, size_t ws_size,
                              hipStream_t stream) {
  const float* x    = (const float*)d_in[0];   // [T, H]
  const float* gout = (const float*)d_in[1];   // [T, E]
  const float* w1   = (const float*)d_in[2];   // [E, 2I, H]
  const float* w2   = (const float*)d_in[3];   // [E, H, I]
  float* out        = (float*)d_out;           // [T, H]

  int*      cnt   = (int*)d_ws;                     // 16 ints
  int*      tmap  = cnt + 16;                       // E*T ints
  float*    wslot = (float*)(tmap + E_ * T_);       // 2*T floats
  _Float16* h_ws  = (_Float16*)(wslot + 2 * T_);    // 2*T * I f16 (~46 MB)

  moe_zero<<<(T_ * H_ + 255) / 256, 256, 0, stream>>>(out, cnt);
  moe_route<<<(T_ + 255) / 256, 256, 0, stream>>>(gout, cnt, tmap, wslot);

  dim3 g1(I_ / BN, (T_ + BM - 1) / BM, E_);   // (44, 32, 8); empty m-tiles exit
  moe_gemm1<<<g1, 256, 0, stream>>>(x, w1, cnt, tmap, h_ws);

  dim3 g2(H_ / BN, (T_ + BM - 1) / BM, E_);   // (16, 32, 8)
  moe_gemm2<<<g2, 256, 0, stream>>>(h_ws, w2, cnt, tmap, wslot, out);
}
